// HomoGraphEncoder_7851200217938
// MI455X (gfx1250) — matrix operand order repaired and checked by reading, not compiled, not run on hardware
//
#include <hip/hip_runtime.h>
#include <hip/hip_bf16.h>

// ---------------------------------------------------------------------------
// GATv2 x2 for MI455X (gfx1250), wave32 + WMMA bf16.
//   N=2048 nodes, 8 heads x 32 dims, hidden 256.
//
// Roofline reasoning (MI455X): all operands are L2-resident (192 MB L2;
// adj=16MB, P=67MB, everything else <8MB), HBM floor ~2us @ 23.3 TB/s.
// The binding cost is the N^2*H softmax elementwise work, so:
//   - exp() is evaluated exactly ONCE per (i,j,h): the stats pass writes
//     unnormalized P = exp(e - rowmax) as bf16 while accumulating Z.
//   - aggregation is then a pure bf16 WMMA GEMM (P @ Wh_head), 1/Z and
//     BN+ReLU folded into the C/D epilogue.
//   - feature transform Wh = X @ W is a bf16 WMMA GEMM as well.
// A recompute-attn fallback (no P tensor) is kept for small workspaces;
// the choice is made from ws_size at launch (deterministic).
// ---------------------------------------------------------------------------

#define NN 2048      // nodes
#define HF 256       // hidden features
#define NH 8         // heads
#define HD 32        // dims per head
#define ALPHA_SLOPE 0.2f
#define BN_RSQRT 0.99999500003749968f   // 1/sqrt(1 + 1e-5)

typedef __attribute__((ext_vector_type(16))) __bf16          v16bf;
typedef __attribute__((ext_vector_type(8)))  float           v8f;
typedef __attribute__((ext_vector_type(16))) unsigned short  us16;
typedef __attribute__((ext_vector_type(8)))  unsigned short  us8;
typedef __attribute__((ext_vector_type(4)))  float           vf4;

__device__ __forceinline__ unsigned short f2bf(float f) {
  unsigned int u = __builtin_bit_cast(unsigned int, f);
  u += 0x7FFFu + ((u >> 16) & 1u);          // round-to-nearest-even
  return (unsigned short)(u >> 16);
}

// ---------------- elementwise converters ----------------------------------
__global__ void k_f32_to_bf16(const float* __restrict__ in,
                              unsigned short* __restrict__ out, int n) {
  int i = blockIdx.x * blockDim.x + threadIdx.x;
  if (i < n) out[i] = f2bf(in[i]);
}

// W [K, 256] f32  ->  WT [256, K] bf16   (B operand wants K contiguous)
__global__ void k_transpose_bf16(const float* __restrict__ W,
                                 unsigned short* __restrict__ WT, int K) {
  int idx = blockIdx.x * blockDim.x + threadIdx.x;
  if (idx >= K * HF) return;
  int k = idx / HF, n = idx % HF;
  WT[n * K + k] = f2bf(W[k * HF + n]);
}

// ---------------- GEMM: Wh = Xbf[2048,K] @ W  (BT = W^T bf16 [256,K]) ------
// One wave -> 16 rows x 32 cols. Also emits WhT bf16 [256,2048] for the
// aggregation pass's B fragments.
__global__ __launch_bounds__(128)
void k_gemm_wmma(const unsigned short* __restrict__ Abf,   // [2048, K] bf16
                 const unsigned short* __restrict__ BT,    // [256, K]  bf16
                 float* __restrict__ Wh,                   // [2048, 256] f32
                 unsigned short* __restrict__ WhT,         // [256, 2048] bf16
                 int K) {
  const int lane  = threadIdx.x & 31;
  const int wid   = blockIdx.x * (blockDim.x >> 5) + (threadIdx.x >> 5);
  const int mtile = wid >> 3;            // 0..127
  const int npair = wid & 7;             // 0..7  (32 cols each)
  const int lrow  = lane & 15;
  const int lhalf = (lane < 16) ? 0 : 1;
  const int row   = mtile * 16 + lrow;
  const int koff1 = lhalf ? 16 : 0;      // A-fragment K runs (contiguous, matches B)
  const int koff2 = lhalf ? 24 : 8;
  const int ncol0 = npair * 32 + lrow;
  const int ncol1 = ncol0 + 16;

  const unsigned short* arow = Abf + (size_t)row * K;
  const unsigned short* b0r  = BT + (size_t)ncol0 * K;
  const unsigned short* b1r  = BT + (size_t)ncol1 * K;

  v8f acc0 = {}; v8f acc1 = {};
  for (int k0 = 0; k0 < K; k0 += 32) {
    us8 alo = *reinterpret_cast<const us8*>(arow + k0 + koff1);
    us8 ahi = *reinterpret_cast<const us8*>(arow + k0 + koff2);
    us16 af;
#pragma unroll
    for (int e = 0; e < 8; ++e) { af[e] = alo[e]; af[8 + e] = ahi[e]; }
    const int kb = k0 + (lhalf ? 16 : 0);
    us16 bf0 = *reinterpret_cast<const us16*>(b0r + kb);
    us16 bf1 = *reinterpret_cast<const us16*>(b1r + kb);
    v16bf av = __builtin_bit_cast(v16bf, af);
    acc0 = __builtin_amdgcn_wmma_f32_16x16x32_bf16(
        false, av, false, __builtin_bit_cast(v16bf, bf0), (short)0, acc0, false, false);
    acc1 = __builtin_amdgcn_wmma_f32_16x16x32_bf16(
        false, av, false, __builtin_bit_cast(v16bf, bf1), (short)0, acc1, false, false);
  }
#pragma unroll
  for (int r = 0; r < 8; ++r) {            // C/D layout: M = r + 8*lhalf
    int m = mtile * 16 + r + lhalf * 8;
    float v0 = acc0[r], v1 = acc1[r];
    Wh[(size_t)m * HF + ncol0] = v0;
    Wh[(size_t)m * HF + ncol1] = v1;
    WhT[(size_t)ncol0 * NN + m] = f2bf(v0);
    WhT[(size_t)ncol1 * NN + m] = f2bf(v1);
  }
}

// ---------------- head scores: sT[h][j] = Wh[j,h,:].a_src ; tT likewise ----
__global__ void k_scores(const float* __restrict__ Wh, const float* __restrict__ a,
                         float* __restrict__ sT, float* __restrict__ tT) {
  int j = blockIdx.x * blockDim.x + threadIdx.x;
  if (j >= NN) return;
#pragma unroll
  for (int h = 0; h < NH; ++h) {
    const float* wrow = Wh + (size_t)j * HF + h * HD;
    float s = 0.f, t = 0.f;
#pragma unroll
    for (int d = 0; d < HD; ++d) {
      s = fmaf(wrow[d], a[d], s);
      t = fmaf(wrow[d], a[HD + d], t);
    }
    sT[h * NN + j] = s;
    tT[h * NN + j] = t;
  }
}

// ---------------- softmax stats per (row i, head): max, 1/sum -------------
// When P != nullptr the second pass also materializes the UNNORMALIZED
// probabilities P[h][i][j] = exp(e - rowmax) in bf16 (exp done ONCE total;
// 1/Z is applied in the aggregation epilogue).
__global__ __launch_bounds__(256)
void k_stats(const float* __restrict__ adj, const float* __restrict__ sT,
             const float* __restrict__ tT, float* __restrict__ mT,
             float* __restrict__ zinvT, unsigned short* __restrict__ P) {
  __shared__ float adjrow[NN];
  __shared__ float red[NH * 256];
  const int i = blockIdx.x, tid = threadIdx.x;
  for (int j = tid; j < NN; j += 256) adjrow[j] = adj[(size_t)i * NN + j];
  __syncthreads();

  float tval[NH], lmax[NH];
#pragma unroll
  for (int h = 0; h < NH; ++h) { tval[h] = tT[h * NN + i]; lmax[h] = -9e15f; }

  for (int j = tid; j < NN; j += 256) {
    if (adjrow[j] > 0.f) {
#pragma unroll
      for (int h = 0; h < NH; ++h) {
        float e = tval[h] + sT[h * NN + j];
        e = (e > 0.f) ? e : ALPHA_SLOPE * e;
        lmax[h] = fmaxf(lmax[h], e);
      }
    }
  }
#pragma unroll
  for (int h = 0; h < NH; ++h) red[h * 256 + tid] = lmax[h];
  __syncthreads();
  for (int s2 = 128; s2 > 0; s2 >>= 1) {
    if (tid < s2)
#pragma unroll
      for (int h = 0; h < NH; ++h)
        red[h * 256 + tid] = fmaxf(red[h * 256 + tid], red[h * 256 + tid + s2]);
    __syncthreads();
  }
  float mfin[NH];
#pragma unroll
  for (int h = 0; h < NH; ++h) mfin[h] = red[h * 256];
  __syncthreads();

  float lsum[NH];
#pragma unroll
  for (int h = 0; h < NH; ++h) lsum[h] = 0.f;
  for (int j = tid; j < NN; j += 256) {
    const bool on = adjrow[j] > 0.f;
#pragma unroll
    for (int h = 0; h < NH; ++h) {
      float p = 0.f;
      if (on) {
        float e = tval[h] + sT[h * NN + j];
        e = (e > 0.f) ? e : ALPHA_SLOPE * e;
        p = __expf(e - mfin[h]);
        lsum[h] += p;
      }
      if (P) P[((size_t)h * NN + i) * NN + j] = f2bf(p);
    }
  }
#pragma unroll
  for (int h = 0; h < NH; ++h) red[h * 256 + tid] = lsum[h];
  __syncthreads();
  for (int s2 = 128; s2 > 0; s2 >>= 1) {
    if (tid < s2)
#pragma unroll
      for (int h = 0; h < NH; ++h)
        red[h * 256 + tid] += red[h * 256 + tid + s2];
    __syncthreads();
  }
  if (tid < NH) {
    mT[tid * NN + i] = mfin[tid];
    zinvT[tid * NN + i] = 1.0f / fmaxf(red[tid * 256], 1e-30f);
  }
}

// ---------------- FAST aggregation: pure bf16 WMMA GEMM over P ------------
// out[i, h*32+d] = relu(BN( (1/Z_ih) * sum_j P[h][i][j] * Wh[j, h*32+d] ))
// One wave -> (16-row tile, head); A = P rows (contiguous bf16),
// B = WhT rows (contiguous bf16). Same fragment maps as k_gemm_wmma.
__global__ __launch_bounds__(128)
void k_agg_gemm(const unsigned short* __restrict__ P,     // [NH][2048][2048] bf16
                const unsigned short* __restrict__ WhT,   // [256][2048] bf16
                const float* __restrict__ zinvT,          // [NH][2048]
                const float* __restrict__ gamma, const float* __restrict__ beta,
                float* __restrict__ outF,                 // [2048][256] f32
                unsigned short* __restrict__ outBf) {     // [2048][256] bf16
  const int lane  = threadIdx.x & 31;
  const int wid   = blockIdx.x * (blockDim.x >> 5) + (threadIdx.x >> 5);
  const int mtile = wid >> 3;            // 0..127
  const int h     = wid & 7;
  const int lrow  = lane & 15;
  const int lhalf = (lane < 16) ? 0 : 1;
  const int koff1 = lhalf ? 16 : 0;
  const int koff2 = lhalf ? 24 : 8;

  const unsigned short* arow = P + ((size_t)h * NN + (mtile * 16 + lrow)) * NN;
  const unsigned short* b0r  = WhT + (size_t)(h * HD + lrow) * NN;
  const unsigned short* b1r  = WhT + (size_t)(h * HD + 16 + lrow) * NN;

  v8f acc0 = {}; v8f acc1 = {};
  for (int j0 = 0; j0 < NN; j0 += 32) {
    us8 alo = *reinterpret_cast<const us8*>(arow + j0 + koff1);
    us8 ahi = *reinterpret_cast<const us8*>(arow + j0 + koff2);
    us16 af;
#pragma unroll
    for (int e = 0; e < 8; ++e) { af[e] = alo[e]; af[8 + e] = ahi[e]; }
    const int kb = j0 + (lhalf ? 16 : 0);
    us16 bf0 = *reinterpret_cast<const us16*>(b0r + kb);
    us16 bf1 = *reinterpret_cast<const us16*>(b1r + kb);
    v16bf av = __builtin_bit_cast(v16bf, af);
    acc0 = __builtin_amdgcn_wmma_f32_16x16x32_bf16(
        false, av, false, __builtin_bit_cast(v16bf, bf0), (short)0, acc0, false, false);
    acc1 = __builtin_amdgcn_wmma_f32_16x16x32_bf16(
        false, av, false, __builtin_bit_cast(v16bf, bf1), (short)0, acc1, false, false);
  }

  const int c0 = h * HD + lrow, c1 = c0 + 16;
  const float g0 = gamma[c0] * BN_RSQRT, be0 = beta[c0];
  const float g1 = gamma[c1] * BN_RSQRT, be1 = beta[c1];
#pragma unroll
  for (int r = 0; r < 8; ++r) {
    int m = mtile * 16 + r + lhalf * 8;
    float zi = zinvT[h * NN + m];
    float v0 = fmaxf(fmaf(acc0[r] * zi, g0, be0), 0.f);
    float v1 = fmaxf(fmaf(acc1[r] * zi, g1, be1), 0.f);
    outF[(size_t)m * HF + c0] = v0;
    outF[(size_t)m * HF + c1] = v1;
    outBf[(size_t)m * HF + c0] = f2bf(v0);
    outBf[(size_t)m * HF + c1] = f2bf(v1);
  }
}

// ---------------- FALLBACK aggregation: recompute attn per lane -----------
__global__ __launch_bounds__(128)
void k_aggregate(const float* __restrict__ adj, const float* __restrict__ sT,
                 const float* __restrict__ tT, const float* __restrict__ mT,
                 const float* __restrict__ zinvT,
                 const unsigned short* __restrict__ WhT,   // [256][2048] bf16
                 const float* __restrict__ gamma, const float* __restrict__ beta,
                 float* __restrict__ outF,                 // [2048][256] f32
                 unsigned short* __restrict__ outBf) {     // [2048][256] bf16
  const int lane  = threadIdx.x & 31;
  const int wid   = blockIdx.x * (blockDim.x >> 5) + (threadIdx.x >> 5);
  const int mtile = wid >> 3;            // 0..127
  const int h     = wid & 7;
  const int lrow  = lane & 15;
  const int lhalf = (lane < 16) ? 0 : 1;
  const int i     = mtile * 16 + lrow;

  const float tv = tT[h * NN + i];
  const float mv = mT[h * NN + i];
  const float zi = zinvT[h * NN + i];
  const int koff1 = lhalf ? 16 : 0;
  const int koff2 = lhalf ? 24 : 8;

  const float* adjrow = adj + (size_t)i * NN;
  const float* srow   = sT + (size_t)h * NN;
  const unsigned short* b0r = WhT + (size_t)(h * HD + lrow) * NN;
  const unsigned short* b1r = WhT + (size_t)(h * HD + 16 + lrow) * NN;

  v8f acc0 = {}; v8f acc1 = {};
  for (int j0 = 0; j0 < NN; j0 += 32) {
    vf4 A[4], S[4];
    A[0] = *reinterpret_cast<const vf4*>(adjrow + j0 + koff1);
    A[1] = *reinterpret_cast<const vf4*>(adjrow + j0 + koff1 + 4);
    A[2] = *reinterpret_cast<const vf4*>(adjrow + j0 + koff2);
    A[3] = *reinterpret_cast<const vf4*>(adjrow + j0 + koff2 + 4);
    S[0] = *reinterpret_cast<const vf4*>(srow + j0 + koff1);
    S[1] = *reinterpret_cast<const vf4*>(srow + j0 + koff1 + 4);
    S[2] = *reinterpret_cast<const vf4*>(srow + j0 + koff2);
    S[3] = *reinterpret_cast<const vf4*>(srow + j0 + koff2 + 4);
    if (j0 + 32 < NN) __builtin_prefetch(adjrow + j0 + 32, 0, 0); // global_prefetch_b8

    us16 af;
#pragma unroll
    for (int q = 0; q < 4; ++q) {
#pragma unroll
      for (int c = 0; c < 4; ++c) {
        float e = tv + S[q][c];
        e = (e > 0.f) ? e : ALPHA_SLOPE * e;
        float p = (A[q][c] > 0.f) ? __expf(e - mv) * zi : 0.f;
        af[q * 4 + c] = f2bf(p);
      }
    }
    const int kb = j0 + (lhalf ? 16 : 0);
    us16 bf0 = *reinterpret_cast<const us16*>(b0r + kb);
    us16 bf1 = *reinterpret_cast<const us16*>(b1r + kb);
    v16bf av = __builtin_bit_cast(v16bf, af);
    acc0 = __builtin_amdgcn_wmma_f32_16x16x32_bf16(
        false, av, false, __builtin_bit_cast(v16bf, bf0), (short)0, acc0, false, false);
    acc1 = __builtin_amdgcn_wmma_f32_16x16x32_bf16(
        false, av, false, __builtin_bit_cast(v16bf, bf1), (short)0, acc1, false, false);
  }

  const int c0 = h * HD + lrow, c1 = c0 + 16;
  const float g0 = gamma[c0] * BN_RSQRT, be0 = beta[c0];
  const float g1 = gamma[c1] * BN_RSQRT, be1 = beta[c1];
#pragma unroll
  for (int r = 0; r < 8; ++r) {
    int m = mtile * 16 + r + lhalf * 8;
    float v0 = fmaxf(fmaf(acc0[r], g0, be0), 0.f);
    float v1 = fmaxf(fmaf(acc1[r], g1, be1), 0.f);
    outF[(size_t)m * HF + c0] = v0;
    outF[(size_t)m * HF + c1] = v1;
    outBf[(size_t)m * HF + c0] = f2bf(v0);
    outBf[(size_t)m * HF + c1] = f2bf(v1);
  }
}

// ---------------------------------------------------------------------------
extern "C" void kernel_launch(void* const* d_in, const int* in_sizes, int n_in,
                              void* d_out, int out_size, void* d_ws, size_t ws_size,
                              hipStream_t stream) {
  const float* x   = (const float*)d_in[0];
  const float* adj = (const float*)d_in[1];
  const float* W0  = (const float*)d_in[2];
  const float* a0  = (const float*)d_in[3];
  const float* g0  = (const float*)d_in[4];
  const float* b0  = (const float*)d_in[5];
  const float* W1  = (const float*)d_in[6];
  const float* a1  = (const float*)d_in[7];
  const float* g1  = (const float*)d_in[8];
  const float* b1  = (const float*)d_in[9];
  float* out = (float*)d_out;

  // workspace carve-up (base ~6.8 MB; +67 MB for the materialized-P path)
  size_t off = 0;
  auto carve = [&](size_t bytes) {
    void* p = (char*)d_ws + off;
    off += (bytes + 255) & ~(size_t)255;
    return p;
  };
  unsigned short* xbf  = (unsigned short*)carve((size_t)NN * 512 * 2); // layer1 A
  unsigned short* w0t  = (unsigned short*)carve((size_t)HF * 512 * 2);
  unsigned short* w1t  = (unsigned short*)carve((size_t)HF * HF * 2);
  float*          wh   = (float*)carve((size_t)NN * HF * 4);
  unsigned short* wht  = (unsigned short*)carve((size_t)HF * NN * 2);
  float*          sT   = (float*)carve((size_t)NH * NN * 4);
  float*          tT   = (float*)carve((size_t)NH * NN * 4);
  float*          mT   = (float*)carve((size_t)NH * NN * 4);
  float*          zT   = (float*)carve((size_t)NH * NN * 4);
  unsigned short* h1bf = (unsigned short*)carve((size_t)NN * HF * 2);
  unsigned short* Pbf  = (unsigned short*)carve((size_t)NH * NN * NN * 2);
  const bool fast = (ws_size >= off);   // deterministic path choice
  unsigned short* Parg = fast ? Pbf : nullptr;

  // ---- precision conversions --------------------------------------------
  k_f32_to_bf16<<<(NN * 512 + 255) / 256, 256, 0, stream>>>(x, xbf, NN * 512);
  k_transpose_bf16<<<(512 * HF + 255) / 256, 256, 0, stream>>>(W0, w0t, 512);
  k_transpose_bf16<<<(HF * HF + 255) / 256, 256, 0, stream>>>(W1, w1t, HF);

  // ---- layer 1 (f32 result staged in d_out, bf16 copy feeds layer 2) ----
  k_gemm_wmma<<<256, 128, 0, stream>>>(xbf, w0t, wh, wht, 512);
  k_scores<<<NN / 256, 256, 0, stream>>>(wh, a0, sT, tT);
  k_stats<<<NN, 256, 0, stream>>>(adj, sT, tT, mT, zT, Parg);
  if (fast)
    k_agg_gemm<<<256, 128, 0, stream>>>(Pbf, wht, zT, g0, b0, out, h1bf);
  else
    k_aggregate<<<256, 128, 0, stream>>>(adj, sT, tT, mT, zT, wht, g0, b0,
                                         out, h1bf);

  // ---- layer 2 (bf16 scratch output reuses xbf region) ------------------
  k_gemm_wmma<<<256, 128, 0, stream>>>(h1bf, w1t, wh, wht, HF);
  k_scores<<<NN / 256, 256, 0, stream>>>(wh, a1, sT, tT);
  k_stats<<<NN, 256, 0, stream>>>(adj, sT, tT, mT, zT, Parg);
  if (fast)
    k_agg_gemm<<<256, 128, 0, stream>>>(Pbf, wht, zT, g1, b1, out, xbf);
  else
    k_aggregate<<<256, 128, 0, stream>>>(adj, sT, tT, mT, zT, wht, g1, b1,
                                         out, xbf);
}
